// dot_product_attention_14826227106481
// MI455X (gfx1250) — matrix operand order repaired and checked by reading, not compiled, not run on hardware
//
#include <hip/hip_runtime.h>
#include <stdint.h>

// ---------------------------------------------------------------------------
// Types for CDNA5 WMMA (wave32): 16x16x32 f16 -> f32 accumulate
// ---------------------------------------------------------------------------
typedef _Float16 h16x16 __attribute__((ext_vector_type(16)));
typedef _Float16 h16x8  __attribute__((ext_vector_type(8)));
typedef float    f32x8  __attribute__((ext_vector_type(8)));

#define WMMA_F16(a, b, c) \
    __builtin_amdgcn_wmma_f32_16x16x32_f16(false, (a), false, (b), (short)0, (c), false, false)

// CDNA5 async global->LDS copy (ASYNCcnt-tracked).  INST_OFFSET is added to
// the global source address only; callers advance the LDS destination
// explicitly, so a run of 16B chunks shares one gptr but not one ldsoff.
#define ASYNC_LD_B128(ldsoff, gptr, imm)                                   \
    asm volatile("global_load_async_to_lds_b128 %0, %1, off offset:%c2"    \
                 :: "v"(ldsoff), "v"(gptr), "i"(imm) : "memory")
#define WAIT_ASYNC(n) \
    asm volatile("s_wait_asynccnt %c0" :: "i"(n) : "memory")

__device__ __forceinline__ unsigned ldsOff(const void* p) {
    // generic pointers to LDS carry the LDS byte offset in their low 32 bits
    return (unsigned)(uintptr_t)p;
}

// Workspace layout (bytes):
//   WT  : f16 [192][1024]      @ 0x000000  (W transposed, K-contig per column)
//   Qs  : f16 [8192][64]       @ 0x060000  (Q pre-scaled by 1/8)
//   Kc  : f16 [8192][64]       @ 0x160000
//   VT  : f16 [4][64][2048]    @ 0x260000  (V transposed: key index contiguous)
#define WS_WT 0x000000
#define WS_Q  0x060000
#define WS_K  0x160000
#define WS_VT 0x260000

// ---------------------------------------------------------------------------
// Kernel 1: transpose W (f32 [1024][192]) -> WT (f16 [192][1024])
// ---------------------------------------------------------------------------
__global__ __launch_bounds__(256) void wt_transpose_kernel(
    const float* __restrict__ W, _Float16* __restrict__ WT) {
    __shared__ _Float16 tile[16][17];
    const int tx = threadIdx.x & 15;
    const int ty = threadIdx.x >> 4;
    const int c0 = (blockIdx.x % 12) * 16;
    const int k0 = (blockIdx.x / 12) * 16;
    tile[ty][tx] = (_Float16)W[(k0 + ty) * 192 + (c0 + tx)];
    __syncthreads();
    WT[(size_t)(c0 + ty) * 1024 + (k0 + tx)] = tile[tx][ty];
}

// ---------------------------------------------------------------------------
// Kernel 2: QKV projection via WMMA with double-buffered async W staging.
// Workgroup = 4 waves x 16-row tiles. Per K-step the 192x32 W slab (12KB,
// identical for all 4 waves) is async-copied into LDS one step ahead.
// B fragments are software-pipelined in registers so ds_load latency hides
// under the previous WMMA.
// ---------------------------------------------------------------------------
__global__ __launch_bounds__(128) void qkv_proj_kernel(
    const float* __restrict__ x, const _Float16* __restrict__ WT,
    const float* __restrict__ bias,
    _Float16* __restrict__ Qs, _Float16* __restrict__ Kc,
    _Float16* __restrict__ VT) {
    __shared__ __align__(16) _Float16 wslab[2][192 * 32];   // 2 x 12KB

    const int lane = threadIdx.x & 31;
    const int wave = threadIdx.x >> 5;
    const int tile = blockIdx.x * 4 + wave;        // 0..511
    const int row0 = tile * 16;
    const int n    = lane & 15;
    const int half = lane >> 4;
    const int tid  = threadIdx.x;

    const float* xrow = x + (size_t)(row0 + (lane & 15)) * 1024;

    // cooperative copy geometry: 768 16B-chunks per slab, 6 per thread
    const int part = tid & 3;                      // 16B chunk within column
    const int col0 = tid >> 2;                     // first column (step 32)
    const _Float16* wsrc0 = WT + (size_t)col0 * 1024 + part * 8;

    // prologue: issue slab 0
    {
        const unsigned dst = ldsOff(&wslab[0][0]) + tid * 16;
#pragma unroll
        for (int j = 0; j < 6; ++j)
            ASYNC_LD_B128(dst + j * 2048, wsrc0 + (size_t)j * 32 * 1024, 0);
    }

    f32x8 acc[12];
#pragma unroll
    for (int ct = 0; ct < 12; ++ct) acc[ct] = f32x8{};

    for (int ks = 0; ks < 32; ++ks) {              // K-steps of 32
        const int k0 = ks * 32;
        WAIT_ASYNC(0);                             // my slab-ks writes done
        __syncthreads();                           // slab valid for all waves
        if (ks + 1 < 32) {                         // issue slab ks+1
            const unsigned dst = ldsOff(&wslab[(ks + 1) & 1][0]) + tid * 16;
            const _Float16* src = wsrc0 + (k0 + 32);
#pragma unroll
            for (int j = 0; j < 6; ++j)
                ASYNC_LD_B128(dst + j * 2048, src + (size_t)j * 32 * 1024, 0);
            // prefetch next x slab while this step computes
            __builtin_prefetch((const void*)(xrow + k0 + 32), 0, 1);
            __builtin_prefetch((const void*)(xrow + k0 + 48), 0, 1);
        }

        // A fragment: lane holds row (lane&15), K = {16h..16h+15}
        h16x16 a;
        const float* p0 = xrow + k0 + half * 16;
        const float* p1 = xrow + k0 + half * 16 + 8;
#pragma unroll
        for (int i = 0; i < 8; ++i) {
            a[i]     = (_Float16)p0[i];
            a[i + 8] = (_Float16)p1[i];
        }
        // 12 B fragments from LDS (layout: [col][32 k-values]),
        // pipelined one fragment ahead in registers
        const _Float16* wb = &wslab[ks & 1][0] + (size_t)n * 32 + half * 16;
        h16x16 bcur = *(const h16x16*)(wb);
#pragma unroll
        for (int ct = 0; ct < 12; ++ct) {
            h16x16 bnext = (ct + 1 < 12)
                ? *(const h16x16*)(wb + (size_t)(ct + 1) * 16 * 32) : bcur;
            acc[ct] = WMMA_F16(a, bcur, acc[ct]);
            bcur = bnext;
        }
        __syncthreads();                           // reads done before overwrite
    }

    // ---- epilogue: uniform routing per column tile ----
    const int  rQ = row0 + half * 8;
    const int  bb = row0 >> 11;
    const int  sb = (row0 & 2047) + half * 8;
#pragma unroll
    for (int ct = 0; ct < 4; ++ct) {               // Q columns 0..63
        const int c = ct * 16 + n;
        const float bv = bias[c];
        _Float16* qp = Qs + (size_t)rQ * 64 + c;
#pragma unroll
        for (int r = 0; r < 8; ++r)
            qp[(size_t)r * 64] = (_Float16)((acc[ct][r] + bv) * 0.125f);
    }
#pragma unroll
    for (int ct = 4; ct < 8; ++ct) {               // K columns 64..127
        const int c = ct * 16 + n;
        const float bv = bias[c];
        _Float16* kp = Kc + (size_t)rQ * 64 + (c - 64);
#pragma unroll
        for (int r = 0; r < 8; ++r)
            kp[(size_t)r * 64] = (_Float16)(acc[ct][r] + bv);
    }
#pragma unroll
    for (int ct = 8; ct < 12; ++ct) {              // V columns 128..191 -> V^T
        const int c = ct * 16 + n;
        const float bv = bias[c];
        h16x8 pk;
#pragma unroll
        for (int r = 0; r < 8; ++r) pk[r] = (_Float16)(acc[ct][r] + bv);
        *(h16x8*)(VT + ((size_t)bb * 64 + (c - 128)) * 2048 + sb) = pk;
    }
}

// ---------------------------------------------------------------------------
// Kernel 3: causal flash attention; per-wave double-buffered async K/V staging.
// Staging per wave/buffer: [0..2047]   = K block, 32 rows x 64 f16 (row-major)
//                          [2048..4095] = V block, 64 d-rows x 32 keys
// All 8 fragment ds_loads are issued up-front so the softmax chain covers the
// V-fragment latency.
// ---------------------------------------------------------------------------
__global__ __launch_bounds__(128) void flash_attn_kernel(
    const _Float16* __restrict__ Qs, const _Float16* __restrict__ Kc,
    const _Float16* __restrict__ VT, float* __restrict__ out) {
    __shared__ __align__(16) _Float16 fstage[4][2][4096];   // 64KB
    __shared__ __align__(16) _Float16 pbuf[4][16 * 32];     // P staging

    const int lane = threadIdx.x & 31;
    const int wave = threadIdx.x >> 5;
    const int tile = blockIdx.x * 4 + wave;        // 0..511
    const int b    = tile >> 7;
    const int qs   = (tile & 127) * 16;
    const int n    = lane & 15;
    const int half = lane >> 4;

    const _Float16* Qb = Qs + (size_t)b * 2048 * 64;
    const _Float16* Kb = Kc + (size_t)b * 2048 * 64;
    const _Float16* Vb = VT + (size_t)b * 64 * 2048;

    // async-stage one 32-key K/V block into fstage[wave][buf]
    auto stage_block = [&](int kb, int buf) {
        const unsigned base = ldsOff(&fstage[wave][buf][0]);
        const _Float16* kg = Kb + (size_t)(kb + lane) * 64;   // K row `lane`
        const unsigned  kd = base + lane * 128;
#pragma unroll
        for (int j = 0; j < 8; ++j) ASYNC_LD_B128(kd + j * 16, kg, j * 16);
        const _Float16* vg0 = Vb + (size_t)lane * 2048 + kb;  // V rows lane,lane+32
        const _Float16* vg1 = vg0 + (size_t)32 * 2048;
        const unsigned  vd0 = base + 4096 + lane * 64;        // bytes
        const unsigned  vd1 = vd0 + 32 * 64;
#pragma unroll
        for (int j = 0; j < 4; ++j) ASYNC_LD_B128(vd0 + j * 16, vg0, j * 16);
#pragma unroll
        for (int j = 0; j < 4; ++j) ASYNC_LD_B128(vd1 + j * 16, vg1, j * 16);
    };

    const int nblocks = (qs + 16 + 31) >> 5;
    stage_block(0, 0);                             // prologue: block 0

    // Q fragments (D=64 -> two K-steps of 32), overlap with async staging
    const _Float16* qrow = Qb + (size_t)(qs + (lane & 15)) * 64;
    h16x16 aq0, aq1;
    {
        h16x8 c0 = *(const h16x8*)(qrow + half * 16);
        h16x8 c1 = *(const h16x8*)(qrow + 8 + half * 16);
        h16x8 c2 = *(const h16x8*)(qrow + 32 + half * 16);
        h16x8 c3 = *(const h16x8*)(qrow + 40 + half * 16);
#pragma unroll
        for (int i = 0; i < 8; ++i) {
            aq0[i] = c0[i]; aq0[i + 8] = c1[i];
            aq1[i] = c2[i]; aq1[i + 8] = c3[i];
        }
    }

    f32x8 acc0 = {}, acc1 = {}, acc2 = {}, acc3 = {};
    float m_run[8], l_run[8];
#pragma unroll
    for (int r = 0; r < 8; ++r) { m_run[r] = -1e30f; l_run[r] = 0.0f; }

    _Float16* pw = pbuf[wave];

    for (int blk = 0; blk < nblocks; ++blk) {
        const int kb = blk * 32;
        if (blk + 1 < nblocks) {
            stage_block(kb + 32, (blk + 1) & 1);   // prefetch next block
            WAIT_ASYNC(16);                        // in-order: block blk landed
        } else {
            WAIT_ASYNC(0);
        }
        const _Float16* kl = &fstage[wave][blk & 1][0];

        // ---- issue ALL fragment loads up front (DS returns in order) ----
        const _Float16* kr0 = kl + (size_t)n * 64 + half * 16;
        const _Float16* kr1 = kr0 + 16 * 64;
        h16x16 bk0lo = *(const h16x16*)(kr0);
        h16x16 bk0hi = *(const h16x16*)(kr0 + 32);
        h16x16 bk1lo = *(const h16x16*)(kr1);
        h16x16 bk1hi = *(const h16x16*)(kr1 + 32);
        const _Float16* vp = kl + 2048 + (size_t)n * 32 + half * 16;
        h16x16 bv0 = *(const h16x16*)(vp);
        h16x16 bv1 = *(const h16x16*)(vp + 16 * 32);
        h16x16 bv2 = *(const h16x16*)(vp + 32 * 32);
        h16x16 bv3 = *(const h16x16*)(vp + 48 * 32);

        // ---- scores: Q @ K^T for 32 keys (two 16x16 tiles) ----
        f32x8 s0 = {}, s1 = {};
        s0 = WMMA_F16(aq0, bk0lo, s0);
        s0 = WMMA_F16(aq1, bk0hi, s0);
        s1 = WMMA_F16(aq0, bk1lo, s1);
        s1 = WMMA_F16(aq1, bk1hi, s1);

        // ---- causal mask + online softmax ----
        const int k0g = kb + n, k1g = kb + 16 + n;
        float alpha[8];
#pragma unroll
        for (int r = 0; r < 8; ++r) {
            const int qg = qs + half * 8 + r;
            if (k0g > qg) s0[r] = -1e30f;
            if (k1g > qg) s1[r] = -1e30f;
            float v = fmaxf(s0[r], s1[r]);
            v = fmaxf(v, __shfl_xor(v, 1, 32));
            v = fmaxf(v, __shfl_xor(v, 2, 32));
            v = fmaxf(v, __shfl_xor(v, 4, 32));
            v = fmaxf(v, __shfl_xor(v, 8, 32));
            const float mnew = fmaxf(m_run[r], v);
            const float p0 = __expf(s0[r] - mnew);
            const float p1 = __expf(s1[r] - mnew);
            float rs = p0 + p1;
            rs += __shfl_xor(rs, 1, 32);
            rs += __shfl_xor(rs, 2, 32);
            rs += __shfl_xor(rs, 4, 32);
            rs += __shfl_xor(rs, 8, 32);
            alpha[r] = __expf(m_run[r] - mnew);
            l_run[r] = l_run[r] * alpha[r] + rs;
            m_run[r] = mnew;
            const int m = half * 8 + r;
            pw[m * 32 + n]      = (_Float16)p0;
            pw[m * 32 + 16 + n] = (_Float16)p1;
        }
        asm volatile("s_wait_dscnt 0" ::: "memory");

        // ---- reload P as a 16x32 A fragment ----
        h16x8 plo = *(const h16x8*)(pw + (lane & 15) * 32 + half * 16);
        h16x8 phi = *(const h16x8*)(pw + (lane & 15) * 32 + 8 + half * 16);
        h16x16 ap;
#pragma unroll
        for (int i = 0; i < 8; ++i) { ap[i] = plo[i]; ap[i + 8] = phi[i]; }
        asm volatile("" ::: "memory");   // keep next iter's DS stores after reads

        // ---- rescale accumulators and add P @ V ----
#pragma unroll
        for (int r = 0; r < 8; ++r) {
            acc0[r] *= alpha[r]; acc1[r] *= alpha[r];
            acc2[r] *= alpha[r]; acc3[r] *= alpha[r];
        }
        acc0 = WMMA_F16(ap, bv0, acc0);
        acc1 = WMMA_F16(ap, bv1, acc1);
        acc2 = WMMA_F16(ap, bv2, acc2);
        acc3 = WMMA_F16(ap, bv3, acc3);
    }

    // ---- normalize and write out [B,S,64] f32 ----
    float* orow = out + ((size_t)b * 2048 + qs) * 64;
#pragma unroll
    for (int r = 0; r < 8; ++r) {
        const float inv = 1.0f / l_run[r];
        const size_t ro = (size_t)(half * 8 + r) * 64;
        orow[ro + n]      = acc0[r] * inv;
        orow[ro + 16 + n] = acc1[r] * inv;
        orow[ro + 32 + n] = acc2[r] * inv;
        orow[ro + 48 + n] = acc3[r] * inv;
    }
}

// ---------------------------------------------------------------------------
extern "C" void kernel_launch(void* const* d_in, const int* in_sizes, int n_in,
                              void* d_out, int out_size, void* d_ws, size_t ws_size,
                              hipStream_t stream) {
    const float* x    = (const float*)d_in[0];   // [4,2048,1024]
    const float* W    = (const float*)d_in[1];   // [1024,192]
    const float* bias = (const float*)d_in[2];   // [192]
    float* out        = (float*)d_out;           // [4,2048,64]

    char* ws = (char*)d_ws;
    _Float16* WT = (_Float16*)(ws + WS_WT);
    _Float16* Qs = (_Float16*)(ws + WS_Q);
    _Float16* Kc = (_Float16*)(ws + WS_K);
    _Float16* VT = (_Float16*)(ws + WS_VT);

    wt_transpose_kernel<<<12 * 64, 256, 0, stream>>>(W, WT);
    qkv_proj_kernel<<<128, 128, 0, stream>>>(x, WT, bias, Qs, Kc, VT);
    flash_attn_kernel<<<128, 128, 0, stream>>>(Qs, Kc, VT, out);
}
